// AutoInt_47794396070449
// MI455X (gfx1250) — matrix operand order repaired and checked by reading, not compiled, not run on hardware
//
#include <hip/hip_runtime.h>
#include <hip/hip_bf16.h>
#include <math.h>

// ---------------------------------------------------------------------------
// AutoInt forward for MI455X (gfx1250, wave32, WMMA f16 16x16x32).
// One workgroup (256 threads = 8 waves) per batch element.
// ---------------------------------------------------------------------------

#define NFEAT  64
#define DIMIN  256
#define NHEAD  8
#define DHEAD  32

typedef __attribute__((ext_vector_type(16))) _Float16 v16h;
typedef __attribute__((ext_vector_type(8)))  _Float16 v8h_t;
typedef __attribute__((ext_vector_type(8)))  float    v8f;

union V16HU { v16h v; v8h_t h[2]; };

// A operand (16x32, f16) from row-major f16 buffer with leading dim `ld`.
// ISA layout: lanes 0-15 hold row M=lane, K = k0+{0..7, 16..23};
//             lanes 16-31 hold row M=lane-16, K = k0+{8..15, 24..31}.
__device__ __forceinline__ v16h load_a16(const _Float16* base, int row0, int k0,
                                         int ld, int lane) {
  const int r   = row0 + (lane & 15);
  const int off = (lane >> 4) << 3;             // 0 or 8
  const _Float16* p = base + r * ld + k0 + off;
  V16HU u;
  u.h[0] = *(const v8h_t*)(p);                  // K = k0+off   .. +7   (16B)
  u.h[1] = *(const v8h_t*)(p + 16);             // K = k0+off+16 .. +23 (16B)
  return u.v;
}

// B operand (32x16, f16) from a TRANSPOSED source: baseT[n][k] row-major,
// leading dim `ld` (so each lane reads two runs of 8 contiguous halves).
// ISA layout: lanes 0-15 hold column N=lane, K = k0+{0..7, 16..23};
//             lanes 16-31 hold column N=lane-16, K = k0+{8..15, 24..31}.
__device__ __forceinline__ v16h load_b16(const _Float16* baseT, int n0, int k0,
                                         int ld, int lane) {
  const int n   = n0 + (lane & 15);
  const int off = (lane >> 4) << 3;             // 0 or 8
  const _Float16* p = baseT + n * ld + k0 + off;
  V16HU u;
  u.h[0] = *(const v8h_t*)(p);
  u.h[1] = *(const v8h_t*)(p + 16);
  return u.v;
}

__device__ __forceinline__ v8f wmma_f16(v16h a, v16h b, v8f c) {
  return __builtin_amdgcn_wmma_f32_16x16x32_f16(
      /*neg_a=*/false, a, /*neg_b=*/false, b,
      /*c_mod=*/(short)0, c, /*reuse_a=*/false, /*reuse_b=*/false);
}

// C/D layout: VGPR r, lanes 0-15 -> (M=m0+r, N=n0+lane); lanes 16-31 -> (M=m0+8+r).
__device__ __forceinline__ void store_cd_f16(_Float16* out, const v8f& c,
                                             int m0, int n0, int ld, int lane) {
  const int m = m0 + ((lane >> 4) << 3);
  const int n = n0 + (lane & 15);
#pragma unroll
  for (int r = 0; r < 8; ++r) out[(m + r) * ld + n] = (_Float16)c[r];
}

// LDS layout (bytes):
//   xh  [64][256] f16 : 0       (32 KB)  -- later overwritten by f = relu(...)
//   qh  [64][256] f16 : 32768   (32 KB)
//   kh  [64][256] f16 : 65536   (32 KB)
//   vT  [256][64] f16 : 98304   (32 KB)  -- V stored transposed
//   zh  [64][256] f16 : 131072  (32 KB)
//   sc  8x[64][64] f32: 163840  (128 KB) -- scores; f16 probs overlaid in-place
//   red [256]     f32 : 294912  (1 KB)
#define SMEM_BYTES (294912 + 1024)

__global__ void __launch_bounds__(256, 1)
autoint_kernel(const float* __restrict__ x,
               const float* __restrict__ bq, const float* __restrict__ bk,
               const float* __restrict__ bv, const float* __restrict__ b1,
               const float* __restrict__ W2, const float* __restrict__ b2,
               const _Float16* __restrict__ wqT, const _Float16* __restrict__ wkT,
               const _Float16* __restrict__ wvT, const _Float16* __restrict__ w1T,
               float* __restrict__ out)
{
  extern __shared__ __align__(16) char smem[];
  _Float16* xh = (_Float16*)(smem + 0);
  _Float16* qh = (_Float16*)(smem + 32768);
  _Float16* kh = (_Float16*)(smem + 65536);
  _Float16* vT = (_Float16*)(smem + 98304);
  _Float16* zh = (_Float16*)(smem + 131072);
  float*    sc = (float*)   (smem + 163840);
  float*    red= (float*)   (smem + 294912);

  const int b    = blockIdx.x;
  const int tid  = threadIdx.x;
  const int lane = tid & 31;      // wave32 on gfx1250
  const int wave = tid >> 5;      // 8 waves

  // ---------------- stage x tile into LDS as f16 (coalesced) ----------------
  const float* xb = x + (size_t)b * (NFEAT * DIMIN);
  for (int i = 0; i < 64; ++i) {
    const int idx = tid + i * 256;
    xh[idx] = (_Float16)xb[idx];
  }
  __syncthreads();

  // ---------------- phase 1: Q/K/V projections (shared A loads) -------------
  for (int t = 0; t < 8; ++t) {
    const int tile = wave * 8 + t;             // 64 tiles, 8 per wave
    const int m0 = (tile >> 4) << 4;           // 0..48
    const int n0 = (tile & 15) << 4;           // 0..240
    v8f cq = {}, ck = {}, cv = {};
    for (int k0 = 0; k0 < DIMIN; k0 += 32) {
      __builtin_prefetch(wqT + (n0 + (lane & 15)) * DIMIN + k0 + 64, 0, 1);
      const v16h a = load_a16(xh, m0, k0, DIMIN, lane);
      cq = wmma_f16(a, load_b16(wqT, n0, k0, DIMIN, lane), cq);
      ck = wmma_f16(a, load_b16(wkT, n0, k0, DIMIN, lane), ck);
      cv = wmma_f16(a, load_b16(wvT, n0, k0, DIMIN, lane), cv);
    }
    const int n    = n0 + (lane & 15);
    const int mrow = m0 + ((lane >> 4) << 3);
    const float bqv = bq[n], bkv = bk[n], bvv = bv[n];
#pragma unroll
    for (int r = 0; r < 8; ++r) {
      qh[(mrow + r) * DIMIN + n] = (_Float16)(cq[r] + bqv);
      kh[(mrow + r) * DIMIN + n] = (_Float16)(ck[r] + bkv);
      vT[n * NFEAT + (mrow + r)] = (_Float16)(cv[r] + bvv);   // transposed V
    }
  }
  __syncthreads();

  // ---------------- phase 2: attention, one head per wave -------------------
  {
    const int h = wave;
    float* shead = sc + h * (NFEAT * NFEAT);
    const float scale = 0.17677669529663687f;  // 1/sqrt(32)

    // S = Q_h * K_h^T  (K rows are contiguous in d => serve directly as B^T)
#pragma unroll
    for (int mt = 0; mt < 4; ++mt) {
#pragma unroll
      for (int nt = 0; nt < 4; ++nt) {
        v8f s = {};
        s = wmma_f16(load_a16(qh, mt * 16, h * DHEAD, DIMIN, lane),
                     load_b16(kh, nt * 16, h * DHEAD, DIMIN, lane), s);
        const int mrow = mt * 16 + ((lane >> 4) << 3);
        const int n    = nt * 16 + (lane & 15);
#pragma unroll
        for (int r = 0; r < 8; ++r)
          shead[(mrow + r) * NFEAT + n] = s[r] * scale;
      }
    }

    // Row softmax (2 rows per lane); write f16 probs in-place over row starts.
    _Float16* ph = (_Float16*)shead;           // row i at halves offset i*128
    for (int rr = 0; rr < 2; ++rr) {
      const int row = lane * 2 + rr;
      float* srow = shead + row * NFEAT;
      float mx = srow[0];
      for (int j = 1; j < NFEAT; ++j) mx = fmaxf(mx, srow[j]);
      float sum = 0.0f;
      for (int j = 0; j < NFEAT; ++j) {
        const float e = __expf(srow[j] - mx);
        srow[j] = e;
        sum += e;
      }
      const float inv = 1.0f / sum;
      _Float16* prow = ph + row * (2 * NFEAT);
      for (int j = 0; j < NFEAT; ++j)          // read f32[j] then clobber word j/2
        prow[j] = (_Float16)(srow[j] * inv);
    }

    // z_h = P * V_h   (A = probs, ld=128 halves; B = vT rows, contiguous in j)
#pragma unroll
    for (int mt = 0; mt < 4; ++mt) {
#pragma unroll
      for (int nt = 0; nt < 2; ++nt) {
        v8f zc = {};
        for (int k0 = 0; k0 < NFEAT; k0 += 32)
          zc = wmma_f16(load_a16(ph, mt * 16, k0, 2 * NFEAT, lane),
                        load_b16(vT, h * DHEAD + nt * 16, k0, NFEAT, lane), zc);
        store_cd_f16(zh, zc, mt * 16, h * DHEAD + nt * 16, DIMIN, lane);
      }
    }
  }
  __syncthreads();

  // ---------------- phase 3: f = relu(z @ W1 + b1 + x), store over xh -------
  for (int t = 0; t < 8; ++t) {
    const int tile = wave * 8 + t;
    const int m0 = (tile >> 4) << 4;
    const int n0 = (tile & 15) << 4;
    v8f cf = {};
    for (int k0 = 0; k0 < DIMIN; k0 += 32) {
      __builtin_prefetch(w1T + (n0 + (lane & 15)) * DIMIN + k0 + 64, 0, 1);
      cf = wmma_f16(load_a16(zh, m0, k0, DIMIN, lane),
                    load_b16(w1T, n0, k0, DIMIN, lane), cf);
    }
    const int n    = n0 + (lane & 15);
    const int mrow = m0 + ((lane >> 4) << 3);
    const float b1v = b1[n];
#pragma unroll
    for (int r = 0; r < 8; ++r) {
      const int idx = (mrow + r) * DIMIN + n;
      const float f = cf[r] + b1v + (float)xh[idx];
      xh[idx] = (_Float16)fmaxf(f, 0.0f);
    }
  }
  __syncthreads();

  // ---------------- phase 4: y = sigmoid(f_flat . W2 + b2) ------------------
  float s = 0.0f;
  for (int i = 0; i < 64; ++i) {
    const int idx = tid * 64 + i;              // matches f.reshape(B, -1) order
    s += (float)xh[idx] * W2[idx];
  }
  red[tid] = s;
  __syncthreads();
  for (int off = 128; off > 0; off >>= 1) {
    if (tid < off) red[tid] += red[tid + off];
    __syncthreads();
  }
  if (tid == 0) {
    const float tval = red[0] + b2[0];
    out[b] = 1.0f / (1.0f + __expf(-tval));
  }
}

// Convert the four 256x256 f32 weight matrices to f16, TRANSPOSED, into d_ws.
__global__ void prep_weights(const float* __restrict__ Wq, const float* __restrict__ Wk,
                             const float* __restrict__ Wv, const float* __restrict__ W1,
                             _Float16* __restrict__ ws) {
  const int idx = blockIdx.x * blockDim.x + threadIdx.x;  // 65536
  const int k = idx >> 8;        // source row
  const int n = idx & 255;       // source col
  const int dst = n * 256 + k;   // transposed
  ws[dst]              = (_Float16)Wq[idx];
  ws[ 65536 + dst]     = (_Float16)Wk[idx];
  ws[131072 + dst]     = (_Float16)Wv[idx];
  ws[196608 + dst]     = (_Float16)W1[idx];
}

extern "C" void kernel_launch(void* const* d_in, const int* in_sizes, int n_in,
                              void* d_out, int out_size, void* d_ws, size_t ws_size,
                              hipStream_t stream) {
  const float* x  = (const float*)d_in[0];
  const float* Wq = (const float*)d_in[1];
  const float* bq = (const float*)d_in[2];
  const float* Wk = (const float*)d_in[3];
  const float* bk = (const float*)d_in[4];
  const float* Wv = (const float*)d_in[5];
  const float* bv = (const float*)d_in[6];
  const float* W1 = (const float*)d_in[7];
  const float* b1 = (const float*)d_in[8];
  const float* W2 = (const float*)d_in[9];
  const float* b2 = (const float*)d_in[10];
  _Float16* wsh = (_Float16*)d_ws;            // needs 4*65536*2 = 512 KB

  prep_weights<<<256, 256, 0, stream>>>(Wq, Wk, Wv, W1, wsh);

  autoint_kernel<<<8192, 256, SMEM_BYTES, stream>>>(
      x, bq, bk, bv, b1, W2, b2,
      wsh, wsh + 65536, wsh + 131072, wsh + 196608,
      (float*)d_out);
}
